// Agent_57896159150265
// MI455X (gfx1250) — matrix operand order repaired and checked by reading, not compiled, not run on hardware
//
#include <hip/hip_runtime.h>
#include <hip/hip_bf16.h>
#include <math.h>

// ---------------- CDNA5 WMMA / TDM types ----------------
typedef __attribute__((ext_vector_type(16))) __bf16 v16bf;
typedef __attribute__((ext_vector_type(8)))  float  v8f;
typedef __attribute__((ext_vector_type(4)))  unsigned int u32x4;
typedef __attribute__((ext_vector_type(8)))  int i32x8;
typedef __attribute__((ext_vector_type(4)))  int i32x4;

#if defined(__has_builtin)
# if __has_builtin(__builtin_amdgcn_tensor_load_to_lds)
#  define USE_TDM 1
# endif
#endif
#ifndef USE_TDM
# define USE_TDM 0
#endif

#define Hc   128
#define Ec   128
#define Pc   64
#define Lc   50
#define NAc  128

// round-to-nearest-even f32 -> bf16 (raw bits)
static __device__ __forceinline__ unsigned short f2bf(float f) {
    union { float f; unsigned u; } a; a.f = f;
    unsigned u = a.u + 0x7FFFu + ((a.u >> 16) & 1u);
    return (unsigned short)(u >> 16);
}

union FragBF { uint4 q[2]; v16bf v; };

// A fragment (16x32 bf16, MxK), A stored [m][k] row-major in LDS.
static __device__ __forceinline__ v16bf load_frag_a(const unsigned short* p) {
    FragBF f;
    f.q[0] = *(const uint4*)(p);        // k .. k+7
    f.q[1] = *(const uint4*)(p + 16);   // k+16 .. k+23
    return f.v;
}
// B fragment (32x16 bf16, KxN), B stored as W[n][k] row-major (k contiguous).
static __device__ __forceinline__ v16bf load_frag_b(const unsigned short* p) {
    FragBF f;
    f.q[0] = *(const uint4*)(p);        // k .. k+7
    f.q[1] = *(const uint4*)(p + 16);   // k+16 .. k+23
    return f.v;
}

#define WMMA_BF16(A, B, C) \
    __builtin_amdgcn_wmma_f32_16x16x32_bf16(false, (A), false, (B), (short)0, (C), false, false)

// ---------------- kernel 0: v = attention @ last_h, zero accumulators ----------------
__global__ void prep_kernel(const float* __restrict__ attention,
                            const float* __restrict__ last_h,
                            float* __restrict__ v, float* __restrict__ acc,
                            float* __restrict__ wsum) {
    int t = threadIdx.x;            // 256 threads, one per row of attention (2H x H)
    float s = 0.f;
    #pragma unroll 4
    for (int j = 0; j < Hc; ++j) s += attention[t * Hc + j] * last_h[j];
    v[t]  = s;
    acc[t] = 0.f;
    if (t == 0) *wsum = 0.f;
}

// ---------------- kernel 0b: convert GRU weights to bf16 (native [n][k] layout) ----------------
__global__ void convert_kernel(const float* __restrict__ W,   // action_W  (256 x 256)
                               const float* __restrict__ TW,  // action_TW (128 x 256)
                               unsigned short* __restrict__ oW,
                               unsigned short* __restrict__ oTW) {
    int i = blockIdx.x * blockDim.x + threadIdx.x;
    if (i < 256 * 256) oW[i]  = f2bf(W[i]);
    if (i < 128 * 256) oTW[i] = f2bf(TW[i]);
}

// ---------------- kernel 1: fused single pass over S (memory-bound, 512 MB) ----------------
// one wave per row: b128 loads, wave32 butterfly -> w_i, accumulate w_i*S_i in registers,
// block-level LDS reduction (ds_add_f32), then one set of global atomics per block.
__global__ void attn_kernel(const float* __restrict__ down,
                            const float* __restrict__ up,
                            const float* __restrict__ v,
                            float* __restrict__ acc, float* __restrict__ wsum,
                            int n) {
    __shared__ float blacc[2 * Hc];
    __shared__ float blw;
    const int tid    = threadIdx.x;
    const int lane   = tid & 31;
    const int wave   = blockIdx.x * (blockDim.x >> 5) + (tid >> 5);
    const int nwaves = gridDim.x * (blockDim.x >> 5);

    blacc[tid] = 0.f;
    if (tid == 0) blw = 0.f;
    __syncthreads();

    const float4 vd = ((const float4*)v)[lane];        // v[lane*4 .. +3]
    const float4 vu = ((const float4*)v)[32 + lane];   // v[128 + lane*4 ..]

    float4 ad = make_float4(0.f, 0.f, 0.f, 0.f);
    float4 au = make_float4(0.f, 0.f, 0.f, 0.f);
    float  wl = 0.f;

    for (int i = wave; i < n; i += nwaves) {
        const float4 d = ((const float4*)(down + (size_t)i * Hc))[lane];
        const float4 u = ((const float4*)(up   + (size_t)i * Hc))[lane];
        __builtin_prefetch((const void*)(down + (size_t)(i + nwaves) * Hc), 0, 0);
        __builtin_prefetch((const void*)(up   + (size_t)(i + nwaves) * Hc), 0, 0);

        float p = d.x * vd.x + d.y * vd.y + d.z * vd.z + d.w * vd.w
                + u.x * vu.x + u.y * vu.y + u.z * vu.z + u.w * vu.w;
        #pragma unroll
        for (int off = 16; off > 0; off >>= 1) p += __shfl_xor(p, off, 32);

        ad.x += p * d.x; ad.y += p * d.y; ad.z += p * d.z; ad.w += p * d.w;
        au.x += p * u.x; au.y += p * u.y; au.z += p * u.z; au.w += p * u.w;
        wl   += p;
    }
    // per-block LDS reduction
    atomicAdd(&blacc[lane * 4 + 0], ad.x);
    atomicAdd(&blacc[lane * 4 + 1], ad.y);
    atomicAdd(&blacc[lane * 4 + 2], ad.z);
    atomicAdd(&blacc[lane * 4 + 3], ad.w);
    atomicAdd(&blacc[Hc + lane * 4 + 0], au.x);
    atomicAdd(&blacc[Hc + lane * 4 + 1], au.y);
    atomicAdd(&blacc[Hc + lane * 4 + 2], au.z);
    atomicAdd(&blacc[Hc + lane * 4 + 3], au.w);
    if (lane == 0) atomicAdd(&blw, wl);
    __syncthreads();
    // one flush per block
    atomicAdd(&acc[tid], blacc[tid]);
    if (tid == 0) atomicAdd(wsum, blw);
}

// ---------------- kernel 2: GRU chain, single WGP, bf16 WMMA, weights in LDS ----------------
// LDS layout (dynamic, 288 KB):
//   sW  : 256*256 bf16 (131072 B)   sTW : 128*256 bf16 (65536 B)
//   sA  : 64*256  bf16 (32768 B)    sH/sZ : 64*128 f32 (32768 B each)
__global__ void __launch_bounds__(256, 1)
chain_kernel(const uint4* __restrict__ gWq,          // bf16 W followed by bf16 TW, contiguous
             const float* __restrict__ act_embed,
             const int*   __restrict__ actions,
             const int*   __restrict__ parent_idx,
             const float* __restrict__ down,
             const float* __restrict__ action_b,
             const float* __restrict__ action_Tb,
             float* __restrict__ pcache) {
    extern __shared__ char smem[];
    unsigned short* sW  = (unsigned short*)smem;
    unsigned short* sTW = sW  + 256 * 256;
    unsigned short* sA  = sTW + 128 * 256;
    float*          sH  = (float*)(sA + 64 * 256);
    float*          sZ  = sH + 64 * 128;

    const int tid   = threadIdx.x;
    const int lane  = tid & 31;
    const int wave  = tid >> 5;
    const int laneN = lane & 15;
    const int hi    = (lane >> 4) & 1;

#if USE_TDM
    // Tensor Data Mover: DMA 196608 B of bf16 weights (24576 x 8B elements) into LDS.
    // 1-D D# descriptor per CDNA5 ISA (data_size=8B, tile_dim0=24576).
    if (tid < 32) {   // one wave issues the DMA (EXEC is ignored by TDM)
        unsigned long long ga = (unsigned long long)(const void*)gWq;
        u32x4 g0;
        g0[0] = 1u;                                        // count=1, user descriptor
        g0[1] = 0u;                                        // lds_addr = dynamic LDS base
        g0[2] = (unsigned)(ga & 0xFFFFFFFFu);              // global_addr[31:0]
        g0[3] = (unsigned)((ga >> 32) & 0x1FFFFFFu)        // global_addr[56:32]
              | (2u << 30);                                // type = 2 ("image")
        i32x8 g1;
        g1[0] = 0x00030000;                // wg_mask=0, data_size=3 (8B), no pad/iterate
        g1[1] = (int)(24576u << 16);       // tensor_dim0[15:0]   (bits 79:48 = 24576)
        g1[2] = (int)(1u << 16);           // tensor_dim0 hi=0 | tensor_dim1=1 (low bits)
        g1[3] = (int)(24576u << 16);       // tensor_dim1 hi=0 | tile_dim0 = 24576
        g1[4] = 0;                         // tile_dim1=0 (unused), tile_dim2=0
        g1[5] = 24576;                     // tensor_dim0_stride (low 32 of 48)
        g1[6] = 0;
        g1[7] = 0;                         // tensor_dim1_stride = 0
        i32x4 z4; z4[0] = 0; z4[1] = 0; z4[2] = 0; z4[3] = 0;
#if __clang_major__ >= 23
        i32x8 z8; for (int i = 0; i < 8; ++i) z8[i] = 0;
        __builtin_amdgcn_tensor_load_to_lds(g0, g1, z4, z4, z8, 0);
#else
        __builtin_amdgcn_tensor_load_to_lds(g0, g1, z4, z4, 0);
#endif
        __builtin_amdgcn_s_wait_tensorcnt(0);
    }
#else
    {
        uint4* dst = (uint4*)smem;
        for (int i = tid; i < 12288; i += 256) dst[i] = gWq[i];
    }
#endif

    // h0 = down_states[parent_idx]   (outside the TDM target region)
    for (int i = tid; i < Pc * Hc; i += 256) {
        int p = i >> 7, j = i & 127;
        sH[i] = down[(size_t)parent_idx[p] * Hc + j];
    }
    __syncthreads();

    for (int l = 0; l < Lc; ++l) {
        // stage A = [ emb(p,l) | h(p) ] in bf16
        for (int i = tid; i < Pc * Hc; i += 256) {
            int p = i >> 7, e = i & 127;
            int a = actions[p * Lc + l];
            sA[p * 256 + e]       = f2bf(act_embed[(size_t)a * Ec + e]);
            sA[p * 256 + 128 + e] = f2bf(sH[i]);
        }
        __syncthreads();

        // ---- GEMM1: (64 x 256) @ W^T(256 x 256) ; per wave: 2 N-tiles x 4 M-tiles
        // loop order: per K-step load 4 A-frags + 2 B-frags once, issue 8 WMMAs
        v8f acc[8];
        for (int ti = 0; ti < 8; ++ti)
            for (int r = 0; r < 8; ++r) acc[ti][r] = 0.f;

        #pragma unroll
        for (int kt = 0; kt < 8; ++kt) {
            const int ka = kt * 32 + (hi ? 8 : 0);
            const int kb = kt * 32 + (hi ? 8 : 0);
            v16bf a0 = load_frag_a(sA + (  0 + laneN) * 256 + ka);
            v16bf a1 = load_frag_a(sA + ( 16 + laneN) * 256 + ka);
            v16bf a2 = load_frag_a(sA + ( 32 + laneN) * 256 + ka);
            v16bf a3 = load_frag_a(sA + ( 48 + laneN) * 256 + ka);
            v16bf b0 = load_frag_b(sW + ((wave * 2 + 0) * 16 + laneN) * 256 + kb);
            v16bf b1 = load_frag_b(sW + ((wave * 2 + 1) * 16 + laneN) * 256 + kb);
            acc[0] = WMMA_BF16(a0, b0, acc[0]);
            acc[1] = WMMA_BF16(a1, b0, acc[1]);
            acc[2] = WMMA_BF16(a2, b0, acc[2]);
            acc[3] = WMMA_BF16(a3, b0, acc[3]);
            acc[4] = WMMA_BF16(a0, b1, acc[4]);
            acc[5] = WMMA_BF16(a1, b1, acc[5]);
            acc[6] = WMMA_BF16(a2, b1, acc[6]);
            acc[7] = WMMA_BF16(a3, b1, acc[7]);
        }
        __syncthreads();   // all A/B reads done before we overwrite sA's h-half

        // sigmoid; r-half -> sA (as bf16 r*h), z-half -> sZ
        for (int ti = 0; ti < 8; ++ti) {
            const int nt = wave * 2 + (ti >> 2);
            const int mt = ti & 3;
            const int n  = nt * 16 + laneN;
            #pragma unroll
            for (int r = 0; r < 8; ++r) {
                const int m = mt * 16 + r + (hi ? 8 : 0);
                float val = acc[ti][r] + action_b[n];
                val = 1.f / (1.f + __expf(-val));
                if (n < 128) sA[m * 256 + 128 + n] = f2bf(val * sH[m * Hc + n]);
                else         sZ[m * Hc + (n - 128)] = val;
            }
        }
        __syncthreads();

        // ---- GEMM2: (64 x 256) @ TW^T(256 x 128) ; per wave: 1 N-tile x 4 M-tiles
        v8f acc2[4];
        for (int ti = 0; ti < 4; ++ti)
            for (int r = 0; r < 8; ++r) acc2[ti][r] = 0.f;

        #pragma unroll
        for (int kt = 0; kt < 8; ++kt) {
            const int ka = kt * 32 + (hi ? 8 : 0);
            const int kb = kt * 32 + (hi ? 8 : 0);
            v16bf a0 = load_frag_a(sA + (  0 + laneN) * 256 + ka);
            v16bf a1 = load_frag_a(sA + ( 16 + laneN) * 256 + ka);
            v16bf a2 = load_frag_a(sA + ( 32 + laneN) * 256 + ka);
            v16bf a3 = load_frag_a(sA + ( 48 + laneN) * 256 + ka);
            v16bf b  = load_frag_b(sTW + (wave * 16 + laneN) * 256 + kb);
            acc2[0] = WMMA_BF16(a0, b, acc2[0]);
            acc2[1] = WMMA_BF16(a1, b, acc2[1]);
            acc2[2] = WMMA_BF16(a2, b, acc2[2]);
            acc2[3] = WMMA_BF16(a3, b, acc2[3]);
        }
        // tanh + gate update; each (m,n) element owned by exactly one lane
        for (int ti = 0; ti < 4; ++ti) {
            const int n = wave * 16 + laneN;
            #pragma unroll
            for (int r = 0; r < 8; ++r) {
                const int m = ti * 16 + r + (hi ? 8 : 0);
                float tv = tanhf(acc2[ti][r] + action_Tb[n]);
                float z  = sZ[m * Hc + n];
                float h  = sH[m * Hc + n];
                sH[m * Hc + n] = (1.f - z) * h + z * tv;
            }
        }
        __syncthreads();   // h fully updated before next staging
    }

    for (int i = tid; i < Pc * Hc; i += 256) pcache[i] = sH[i];
}

// ---------------- kernel 3: heads + softmax (tiny) ----------------
__global__ void final_kernel(const float* __restrict__ down, const float* __restrict__ up,
                             const float* __restrict__ pcg,
                             const float* __restrict__ acc, const float* __restrict__ wsump,
                             const float* __restrict__ schedW, const float* __restrict__ schedWb,
                             const float* __restrict__ schedU, const float* __restrict__ schedUb,
                             const float* __restrict__ schedT, const float* __restrict__ schedTb,
                             const float* __restrict__ out1W, const float* __restrict__ out1b,
                             const float* __restrict__ out2W, const float* __restrict__ out2b,
                             const int*   __restrict__ opidx, float* __restrict__ out) {
    __shared__ float pc[Pc * Hc];
    __shared__ float xv[256], hm[128], rr[128], zz[128], xT[128], rg[128];
    __shared__ float st[384], tm[128], lg[128], red[2];
    const int t = threadIdx.x;      // 256 threads

    for (int i = t; i < Pc * Hc; i += 256) pc[i] = pcg[i];
    const int op = opidx[0];
    xv[t] = (t < 128) ? down[(size_t)op * Hc + t] : up[(size_t)op * Hc + (t - 128)];
    if (t < 128) rg[t] = 0.f;
    __syncthreads();

    if (t < 128) {
        float s = 0.f;
        for (int p = 0; p < Pc; ++p) s += pc[p * Hc + t];
        hm[t] = s * (1.f / (float)Pc);
    }
    __syncthreads();

    if (t < 128) {
        float sr = schedWb[t], sz = schedUb[t], sx = schedTb[t];
        for (int j = 0; j < 256; ++j) {
            float xj = xv[j];
            sr += xj * schedW[t * 384 + j];
            sz += xj * schedU[t * 384 + j];
            sx += xj * schedT[t * 384 + j];
        }
        for (int j = 0; j < 128; ++j) {
            float hj = hm[j];
            sr += hj * schedW[t * 384 + 256 + j];
            sz += hj * schedU[t * 384 + 256 + j];
        }
        rr[t] = 1.f / (1.f + __expf(-sr));
        zz[t] = 1.f / (1.f + __expf(-sz));
        xT[t] = sx;
    }
    __syncthreads();

    for (int idx = t; idx < Pc * Hc; idx += 256) {
        const int p = idx >> 7, i = idx & 127;
        float s = xT[i];
        for (int j = 0; j < 128; ++j) s += rr[j] * pc[p * Hc + j] * schedT[i * 384 + 256 + j];
        float tr = tanhf(s);
        float c = ((1.f - zz[i]) * pc[p * Hc + i] + zz[i] * tr) * (1.f / (float)Pc);
        atomicAdd(&rg[i], c);   // ds_add_f32
    }
    __syncthreads();

    const float invw = 1.f / wsump[0];
    st[t] = acc[t] * invw;                 // attention_state (256)
    if (t < 128) st[256 + t] = rg[t];      // region (128)
    __syncthreads();

    if (t < 128) {
        float s = out1b[t];
        for (int j = 0; j < 384; ++j) s += st[j] * out1W[t * 384 + j];
        tm[t] = s;
    }
    __syncthreads();
    if (t < 128) {
        float s = out2b[t];
        for (int j = 0; j < 128; ++j) s += tm[j] * out2W[t * 128 + j];
        lg[t] = s;
    }
    __syncthreads();
    if (t == 0) {
        float mx = lg[0];
        for (int j = 1; j < NAc; ++j) mx = fmaxf(mx, lg[j]);
        float s = 0.f;
        for (int j = 0; j < NAc; ++j) s += __expf(lg[j] - mx);
        red[0] = mx; red[1] = 1.f / s;
    }
    __syncthreads();
    if (t < NAc) out[t] = __expf(lg[t] - red[0]) * red[1];
}

// ---------------- launcher ----------------
extern "C" void kernel_launch(void* const* d_in, const int* in_sizes, int n_in,
                              void* d_out, int out_size, void* d_ws, size_t ws_size,
                              hipStream_t stream) {
    const float* down       = (const float*)d_in[0];
    const float* up         = (const float*)d_in[1];
    const float* last_h     = (const float*)d_in[2];
    const float* attention  = (const float*)d_in[3];
    const float* act_embed  = (const float*)d_in[4];
    const float* action_W   = (const float*)d_in[5];
    const float* action_b   = (const float*)d_in[6];
    const float* action_TW  = (const float*)d_in[7];
    const float* action_Tb  = (const float*)d_in[8];
    const float* sched_W    = (const float*)d_in[9];
    const float* sched_Wb   = (const float*)d_in[10];
    const float* sched_U    = (const float*)d_in[11];
    const float* sched_Ub   = (const float*)d_in[12];
    const float* sched_T    = (const float*)d_in[13];
    const float* sched_Tb   = (const float*)d_in[14];
    const float* out1_W     = (const float*)d_in[15];
    const float* out1_b     = (const float*)d_in[16];
    const float* out2_W     = (const float*)d_in[17];
    const float* out2_b     = (const float*)d_in[18];
    const int*   actions    = (const int*)d_in[19];
    const int*   parent_idx = (const int*)d_in[20];
    const int*   op_idx     = (const int*)d_in[21];

    const int n = in_sizes[0] / Hc;   // N rows

    // workspace layout
    char* ws = (char*)d_ws;
    float*          v      = (float*)(ws + 0);                    // 256 f
    float*          acc    = (float*)(ws + 1024);                 // 256 f
    float*          wsum   = (float*)(ws + 2048);                 // 1 f
    unsigned short* wsW    = (unsigned short*)(ws + 4096);        // 256*256 bf16
    unsigned short* wsTW   = (unsigned short*)(ws + 4096 + 131072);
    float*          pcache = (float*)(ws + 4096 + 131072 + 65536);

    prep_kernel<<<1, 256, 0, stream>>>(attention, last_h, v, acc, wsum);
    convert_kernel<<<256, 256, 0, stream>>>(action_W, action_TW, wsW, wsTW);
    attn_kernel<<<2048, 256, 0, stream>>>(down, up, v, acc, wsum, n);

    const size_t chain_lds = (size_t)(256 * 256 + 128 * 256 + 64 * 256) * 2  // bf16
                           + (size_t)(64 * 128) * 4 * 2;                     // sH + sZ
    chain_kernel<<<1, 256, chain_lds, stream>>>((const uint4*)wsW, act_embed, actions,
                                                parent_idx, down, action_b, action_Tb, pcache);

    final_kernel<<<1, 256, 0, stream>>>(down, up, pcache, acc, wsum,
                                        sched_W, sched_Wb, sched_U, sched_Ub,
                                        sched_T, sched_Tb, out1_W, out1_b,
                                        out2_W, out2_b, op_idx, (float*)d_out);
}
